// AtomBlock_33303176413270
// MI455X (gfx1250) — compile-verified, matrix operand order checked by reading, not run on hardware
//
#include <hip/hip_runtime.h>

#define DA   128
#define DM   512
#define NAT  4096
#define NT   1024
#define NH   4
#define DH   32
#define WIN  16
#define PP   16384
#define BB   2
#define MTOT (BB * NAT)   // 8192 rows total
#define EPSL 1e-5f

typedef __attribute__((ext_vector_type(16))) __bf16 v16bf;
typedef __attribute__((ext_vector_type(8)))  float  v8f;

// ---------------------------------------------------------------------------
// helpers
// ---------------------------------------------------------------------------
__device__ __forceinline__ float sigmoidf_(float x) {
  return 1.0f / (1.0f + __expf(-x));
}

__device__ __forceinline__ float wave_sum(float v) {
#pragma unroll
  for (int off = 16; off > 0; off >>= 1) v += __shfl_xor(v, off, 32);
  return v;
}

// Load a 16x32 bf16 fragment (A-layout; B uses the mirrored layout which is
// identical when the weight is stored row-major as (n_out, k)).
// lanebase must already include  + (lane&15)*ld + (lane>>4)*8.
__device__ __forceinline__ v16bf frag_ld(const float* __restrict__ lanebase, int k0) {
  v16bf f;
#pragma unroll
  for (int e = 0; e < 8; ++e) f[e] = (__bf16)lanebase[k0 + e];
#pragma unroll
  for (int e = 0; e < 8; ++e) f[e + 8] = (__bf16)lanebase[k0 + 16 + e];
  return f;
}

__device__ __forceinline__ v8f wmma_bf16(v16bf a, v16bf b, v8f c) {
  return __builtin_amdgcn_wmma_f32_16x16x32_bf16(false, a, false, b, (short)0, c,
                                                 false, false);
}

// ---------------------------------------------------------------------------
// zero a float buffer
// ---------------------------------------------------------------------------
__global__ void zero_kernel(float* __restrict__ p, int n) {
  int t = blockIdx.x * blockDim.x + threadIdx.x;
  if (t < n) p[t] = 0.0f;
}

// ---------------------------------------------------------------------------
// scatter pair bias: LN(p_lm)*w+b -> @pair_bias_w^T -> windowed dense buffer
// biasws layout: [b][h][i][j-i+16], stride 33
// ---------------------------------------------------------------------------
__global__ void pair_scatter_kernel(const float* __restrict__ plm,
                                    const int* __restrict__ pidx,
                                    const float* __restrict__ pvm,
                                    const float* __restrict__ lnw,
                                    const float* __restrict__ lnb,
                                    const float* __restrict__ pbw,
                                    float* __restrict__ biasws) {
  int t = blockIdx.x * blockDim.x + threadIdx.x;
  if (t >= BB * PP) return;
  int b = t / PP;
  if (pvm[t] <= 0.0f) return;
  int i = pidx[2 * t + 0];
  int j = pidx[2 * t + 1];
  int d = j - i;
  if (d < -WIN || d > WIN) return;
  const float* x = plm + (size_t)t * 16;
  float mu = 0.0f;
#pragma unroll
  for (int e = 0; e < 16; ++e) mu += x[e];
  mu *= (1.0f / 16.0f);
  float var = 0.0f;
#pragma unroll
  for (int e = 0; e < 16; ++e) { float dd = x[e] - mu; var += dd * dd; }
  var *= (1.0f / 16.0f);
  float rstd = rsqrtf(var + EPSL);
  float y[16];
#pragma unroll
  for (int e = 0; e < 16; ++e) y[e] = (x[e] - mu) * rstd * lnw[e] + lnb[e];
#pragma unroll
  for (int h = 0; h < NH; ++h) {
    float acc = 0.0f;
#pragma unroll
    for (int e = 0; e < 16; ++e) acc += y[e] * pbw[h * 16 + e];
    biasws[(((size_t)b * NH + h) * NAT + i) * 33 + (d + WIN)] = acc;
  }
}

// ---------------------------------------------------------------------------
// cond = t_emb + gather(h_cond, token_idx) @ cond_w^T + cond_b
// grid (M/16, 2), block 128 (4 waves, each one 16x16 tile of the 128 cols)
// ---------------------------------------------------------------------------
__global__ void cond_kernel(const float* __restrict__ hcond,
                            const int* __restrict__ tokidx,
                            const float* __restrict__ condw,
                            const float* __restrict__ condb,
                            const float* __restrict__ temb,
                            float* __restrict__ cond) {
  int wave = threadIdx.x >> 5, lane = threadIdx.x & 31;
  int r = lane & 15, hi = lane >> 4;
  int m0 = blockIdx.x * 16;
  int n0 = blockIdx.y * 64 + wave * 16;
  int m = m0 + r;
  int b = m >> 12;
  int a = m & (NAT - 1);
  const float* arow =
      hcond + ((size_t)b * NT + tokidx[(size_t)b * NAT + a]) * DM + hi * 8;
  const float* brow = condw + (size_t)(n0 + r) * DM + hi * 8;
  v8f acc = {};
#pragma unroll
  for (int k0 = 0; k0 < DM; k0 += 32)
    acc = wmma_bf16(frag_ld(arow, k0), frag_ld(brow, k0), acc);
  int n = n0 + (lane & 15);
#pragma unroll
  for (int e = 0; e < 8; ++e) {
    int mm = m0 + hi * 8 + e;
    int bb_ = mm >> 12;
    cond[(size_t)mm * DA + n] = acc[e] + temb[bb_ * DA + n] + condb[n];
  }
}

// ---------------------------------------------------------------------------
// LayerNorm over rows of 128; one wave per row, writes y*w1 (and y*w2).
// w1/w2 may be null (no weight); o2 may be null (skip second output).
// grid M/8, block 256
// ---------------------------------------------------------------------------
__global__ void ln_dual_kernel(const float* __restrict__ x,
                               const float* __restrict__ w1,
                               const float* __restrict__ w2,
                               float* __restrict__ o1,
                               float* __restrict__ o2) {
  int wave = threadIdx.x >> 5, lane = threadIdx.x & 31;
  int row = blockIdx.x * 8 + wave;
  const float* xr = x + (size_t)row * DA;
  float v[4];
  float s = 0.0f;
#pragma unroll
  for (int i = 0; i < 4; ++i) { v[i] = xr[lane * 4 + i]; s += v[i]; }
  s = wave_sum(s);
  float mu = s * (1.0f / DA);
  float qv = 0.0f;
#pragma unroll
  for (int i = 0; i < 4; ++i) { float d = v[i] - mu; qv += d * d; }
  qv = wave_sum(qv);
  float rstd = rsqrtf(qv * (1.0f / DA) + EPSL);
#pragma unroll
  for (int i = 0; i < 4; ++i) {
    int n = lane * 4 + i;
    float y = (v[i] - mu) * rstd;
    o1[(size_t)row * DA + n] = w1 ? y * w1[n] : y;
    if (o2) o2[(size_t)row * DA + n] = w2 ? y * w2[n] : y;
  }
}

// ---------------------------------------------------------------------------
// AdaLN: out = sigmoid(sn@scw^T + scb) * an + sn@biw^T
// grid M/16, block 256 (8 waves cover 128 cols); dual accumulators per wave
// ---------------------------------------------------------------------------
__global__ void adaln_kernel(const float* __restrict__ sn,
                             const float* __restrict__ an,
                             const float* __restrict__ scw,
                             const float* __restrict__ scb,
                             const float* __restrict__ biw,
                             float* __restrict__ out) {
  int wave = threadIdx.x >> 5, lane = threadIdx.x & 31;
  int r = lane & 15, hi = lane >> 4;
  int m0 = blockIdx.x * 16;
  int n0 = wave * 16;
  const float* arow = sn + (size_t)(m0 + r) * DA + hi * 8;
  const float* b1 = scw + (size_t)(n0 + r) * DA + hi * 8;
  const float* b2 = biw + (size_t)(n0 + r) * DA + hi * 8;
  v8f accS = {}, accB = {};
#pragma unroll
  for (int k0 = 0; k0 < DA; k0 += 32) {
    v16bf af = frag_ld(arow, k0);
    accS = wmma_bf16(af, frag_ld(b1, k0), accS);
    accB = wmma_bf16(af, frag_ld(b2, k0), accB);
  }
  int n = n0 + (lane & 15);
#pragma unroll
  for (int e = 0; e < 8; ++e) {
    int m = m0 + hi * 8 + e;
    float sc = sigmoidf_(accS[e] + scb[n]);
    out[(size_t)m * DA + n] = sc * an[(size_t)m * DA + n] + accB[e];
  }
}

// ---------------------------------------------------------------------------
// gates: g1 = sigmoid(cond@w1^T+b1), g2 = sigmoid(cond@w2^T+b2)
// ---------------------------------------------------------------------------
__global__ void gates_kernel(const float* __restrict__ cond,
                             const float* __restrict__ w1,
                             const float* __restrict__ b1v,
                             const float* __restrict__ w2,
                             const float* __restrict__ b2v,
                             float* __restrict__ g1, float* __restrict__ g2) {
  int wave = threadIdx.x >> 5, lane = threadIdx.x & 31;
  int r = lane & 15, hi = lane >> 4;
  int m0 = blockIdx.x * 16;
  int n0 = wave * 16;
  const float* arow = cond + (size_t)(m0 + r) * DA + hi * 8;
  const float* bw1 = w1 + (size_t)(n0 + r) * DA + hi * 8;
  const float* bw2 = w2 + (size_t)(n0 + r) * DA + hi * 8;
  v8f a1 = {}, a2 = {};
#pragma unroll
  for (int k0 = 0; k0 < DA; k0 += 32) {
    v16bf af = frag_ld(arow, k0);
    a1 = wmma_bf16(af, frag_ld(bw1, k0), a1);
    a2 = wmma_bf16(af, frag_ld(bw2, k0), a2);
  }
  int n = n0 + (lane & 15);
#pragma unroll
  for (int e = 0; e < 8; ++e) {
    int m = m0 + hi * 8 + e;
    g1[(size_t)m * DA + n] = sigmoidf_(a1[e] + b1v[n]);
    g2[(size_t)m * DA + n] = sigmoidf_(a2[e] + b2v[n]);
  }
}

// ---------------------------------------------------------------------------
// Q/K/V/G projections: quad accumulators, one A-fragment per K chunk
// ---------------------------------------------------------------------------
__global__ void qkvg_kernel(const float* __restrict__ qn,
                            const float* __restrict__ wq, const float* __restrict__ qb,
                            const float* __restrict__ wk, const float* __restrict__ wv,
                            const float* __restrict__ wg,
                            float* __restrict__ Q, float* __restrict__ K,
                            float* __restrict__ V, float* __restrict__ G) {
  int wave = threadIdx.x >> 5, lane = threadIdx.x & 31;
  int r = lane & 15, hi = lane >> 4;
  int m0 = blockIdx.x * 16;
  int n0 = wave * 16;
  const float* arow = qn + (size_t)(m0 + r) * DA + hi * 8;
  const float* bq = wq + (size_t)(n0 + r) * DA + hi * 8;
  const float* bk = wk + (size_t)(n0 + r) * DA + hi * 8;
  const float* bv = wv + (size_t)(n0 + r) * DA + hi * 8;
  const float* bg = wg + (size_t)(n0 + r) * DA + hi * 8;
  v8f aQ = {}, aK = {}, aV = {}, aG = {};
#pragma unroll
  for (int k0 = 0; k0 < DA; k0 += 32) {
    v16bf af = frag_ld(arow, k0);
    aQ = wmma_bf16(af, frag_ld(bq, k0), aQ);
    aK = wmma_bf16(af, frag_ld(bk, k0), aK);
    aV = wmma_bf16(af, frag_ld(bv, k0), aV);
    aG = wmma_bf16(af, frag_ld(bg, k0), aG);
  }
  int n = n0 + (lane & 15);
#pragma unroll
  for (int e = 0; e < 8; ++e) {
    size_t o = (size_t)(m0 + hi * 8 + e) * DA + n;
    Q[o] = aQ[e] + qb[n];
    K[o] = aK[e];
    V[o] = aV[e];
    G[o] = aG[e];
  }
}

// ---------------------------------------------------------------------------
// windowed attention, online softmax; one thread per (atom, head)
// ao = sigmoid(G) * att_out      grid (NAT/64, B), block 256
// ---------------------------------------------------------------------------
__global__ void attn_kernel(const float* __restrict__ Q, const float* __restrict__ K,
                            const float* __restrict__ V, const float* __restrict__ G,
                            const float* __restrict__ biasws,
                            const float* __restrict__ padmask,
                            float* __restrict__ ao) {
  int h = threadIdx.x & 3;
  int ir = threadIdx.x >> 2;
  int i = blockIdx.x * 64 + ir;
  int b = blockIdx.y;
  int m = b * NAT + i;
  const float* qrow = Q + (size_t)m * DA + h * DH;
  float qv[DH];
#pragma unroll
  for (int d = 0; d < DH; ++d) qv[d] = qrow[d];
  const float* brow = biasws + (((size_t)b * NH + h) * NAT + i) * 33;
  int j0 = (i - WIN < 0) ? 0 : i - WIN;
  int j1 = (i + WIN > NAT - 1) ? NAT - 1 : i + WIN;
  float mrun = -3.0e38f, denom = 0.0f;
  float acc[DH];
#pragma unroll
  for (int d = 0; d < DH; ++d) acc[d] = 0.0f;
  for (int j = j0; j <= j1; ++j) {
    const float* krow = K + ((size_t)b * NAT + j) * DA + h * DH;
    float s = 0.0f;
#pragma unroll
    for (int d = 0; d < DH; ++d) s += qv[d] * krow[d];
    s *= 0.17677669529663687f;  // 1/sqrt(32)
    s += brow[j - i + WIN];
    s += (1.0f - padmask[(size_t)b * NAT + j]) * -1.0e9f;
    float mnew = fmaxf(mrun, s);
    float corr = __expf(mrun - mnew);
    float w = __expf(s - mnew);
    denom = denom * corr + w;
    const float* vrow = V + ((size_t)b * NAT + j) * DA + h * DH;
#pragma unroll
    for (int d = 0; d < DH; ++d) acc[d] = acc[d] * corr + w * vrow[d];
    mrun = mnew;
  }
  float inv = 1.0f / denom;
  const float* grow = G + (size_t)m * DA + h * DH;
  float* orow = ao + (size_t)m * DA + h * DH;
#pragma unroll
  for (int d = 0; d < DH; ++d) orow[d] = sigmoidf_(grow[d]) * acc[d] * inv;
}

// ---------------------------------------------------------------------------
// q2 = q + gA * (ao @ wo^T)
// ---------------------------------------------------------------------------
__global__ void wo_kernel(const float* __restrict__ ao, const float* __restrict__ wo,
                          const float* __restrict__ qin, const float* __restrict__ gA,
                          float* __restrict__ q2) {
  int wave = threadIdx.x >> 5, lane = threadIdx.x & 31;
  int r = lane & 15, hi = lane >> 4;
  int m0 = blockIdx.x * 16;
  int n0 = wave * 16;
  const float* arow = ao + (size_t)(m0 + r) * DA + hi * 8;
  const float* brow = wo + (size_t)(n0 + r) * DA + hi * 8;
  v8f acc = {};
#pragma unroll
  for (int k0 = 0; k0 < DA; k0 += 32)
    acc = wmma_bf16(frag_ld(arow, k0), frag_ld(brow, k0), acc);
  int n = n0 + (lane & 15);
#pragma unroll
  for (int e = 0; e < 8; ++e) {
    size_t o = (size_t)(m0 + hi * 8 + e) * DA + n;
    q2[o] = qin[o] + gA[o] * acc[e];
  }
}

// ---------------------------------------------------------------------------
// hid = silu(qn2 @ w1^T) * (qn2 @ w3^T)      grid (M/16, 4), block 256
// ---------------------------------------------------------------------------
__global__ void swiglu1_kernel(const float* __restrict__ qn2,
                               const float* __restrict__ w1,
                               const float* __restrict__ w3,
                               float* __restrict__ hid) {
  int wave = threadIdx.x >> 5, lane = threadIdx.x & 31;
  int r = lane & 15, hi = lane >> 4;
  int m0 = blockIdx.x * 16;
  int n0 = blockIdx.y * 128 + wave * 16;
  const float* arow = qn2 + (size_t)(m0 + r) * DA + hi * 8;
  const float* b1 = w1 + (size_t)(n0 + r) * DA + hi * 8;
  const float* b3 = w3 + (size_t)(n0 + r) * DA + hi * 8;
  v8f a1 = {}, a3 = {};
#pragma unroll
  for (int k0 = 0; k0 < DA; k0 += 32) {
    v16bf af = frag_ld(arow, k0);
    a1 = wmma_bf16(af, frag_ld(b1, k0), a1);
    a3 = wmma_bf16(af, frag_ld(b3, k0), a3);
  }
  int n = n0 + (lane & 15);
#pragma unroll
  for (int e = 0; e < 8; ++e) {
    int m = m0 + hi * 8 + e;
    float x = a1[e];
    hid[(size_t)m * DM + n] = (x * sigmoidf_(x)) * a3[e];
  }
}

// ---------------------------------------------------------------------------
// out = (q2 + gT * (hid @ w2^T)) * padmask
// ---------------------------------------------------------------------------
__global__ void final_kernel(const float* __restrict__ hid,
                             const float* __restrict__ w2,
                             const float* __restrict__ q2,
                             const float* __restrict__ gT,
                             const float* __restrict__ padmask,
                             float* __restrict__ out) {
  int wave = threadIdx.x >> 5, lane = threadIdx.x & 31;
  int r = lane & 15, hi = lane >> 4;
  int m0 = blockIdx.x * 16;
  int n0 = wave * 16;
  const float* arow = hid + (size_t)(m0 + r) * DM + hi * 8;
  const float* brow = w2 + (size_t)(n0 + r) * DM + hi * 8;
  v8f acc = {};
#pragma unroll
  for (int k0 = 0; k0 < DM; k0 += 32)
    acc = wmma_bf16(frag_ld(arow, k0), frag_ld(brow, k0), acc);
  int n = n0 + (lane & 15);
#pragma unroll
  for (int e = 0; e < 8; ++e) {
    int m = m0 + hi * 8 + e;
    size_t o = (size_t)m * DA + n;
    out[o] = (q2[o] + gT[o] * acc[e]) * padmask[m];
  }
}

// ---------------------------------------------------------------------------
// host launcher
// ---------------------------------------------------------------------------
extern "C" void kernel_launch(void* const* d_in, const int* in_sizes, int n_in,
                              void* d_out, int out_size, void* d_ws, size_t ws_size,
                              hipStream_t stream) {
  (void)in_sizes; (void)n_in; (void)out_size; (void)ws_size;
  const float* q      = (const float*)d_in[0];
  const float* h_cond = (const float*)d_in[2];
  const float* p_lm   = (const float*)d_in[3];
  const int*   p_idx  = (const int*)d_in[4];
  const float* t_emb  = (const float*)d_in[5];
  const int*   tok    = (const int*)d_in[6];
  const float* pmask  = (const float*)d_in[7];
  const float* pvmask = (const float*)d_in[8];
  const float* cond_w = (const float*)d_in[9];
  const float* cond_b = (const float*)d_in[10];
  const float* a1_snw = (const float*)d_in[11];
  const float* a1_scw = (const float*)d_in[12];
  const float* a1_scb = (const float*)d_in[13];
  const float* a1_biw = (const float*)d_in[14];
  const float* a2_snw = (const float*)d_in[15];
  const float* a2_scw = (const float*)d_in[16];
  const float* a2_scb = (const float*)d_in[17];
  const float* a2_biw = (const float*)d_in[18];
  const float* wq_w   = (const float*)d_in[19];
  const float* wq_b   = (const float*)d_in[20];
  const float* wk_w   = (const float*)d_in[21];
  const float* wv_w   = (const float*)d_in[22];
  const float* wg_w   = (const float*)d_in[23];
  const float* wo_w   = (const float*)d_in[24];
  const float* lnp_w  = (const float*)d_in[25];
  const float* lnp_b  = (const float*)d_in[26];
  const float* pb_w   = (const float*)d_in[27];
  const float* ag_w   = (const float*)d_in[28];
  const float* ag_b   = (const float*)d_in[29];
  const float* tg_w   = (const float*)d_in[30];
  const float* tg_b   = (const float*)d_in[31];
  const float* sw1    = (const float*)d_in[32];
  const float* sw3    = (const float*)d_in[33];
  const float* sw2    = (const float*)d_in[34];
  float* out = (float*)d_out;

  float* ws = (float*)d_ws;
  const size_t NB = (size_t)MTOT * DA;  // 1M floats per (M,128) buffer
  float* cond = ws + 0 * NB;            // later reused as q2
  float* sn1  = ws + 1 * NB;            // later reused as ao
  float* sn2  = ws + 2 * NB;
  float* an   = ws + 3 * NB;            // reused as an2
  float* qn   = ws + 4 * NB;            // reused as qn2
  float* gA   = ws + 5 * NB;
  float* gT   = ws + 6 * NB;
  float* Qb   = ws + 7 * NB;            // 7..10 later reused as hid (4*NB)
  float* Kb   = ws + 8 * NB;
  float* Vb   = ws + 9 * NB;
  float* Gb   = ws + 10 * NB;
  float* bias = ws + 11 * NB;           // B*H*NAT*33 floats
  float* ao   = sn1;
  float* q2   = cond;
  float* an2  = an;
  float* qn2  = qn;
  float* hid  = Qb;

  const int nbias = BB * NH * NAT * 33;
  zero_kernel<<<(nbias + 255) / 256, 256, 0, stream>>>(bias, nbias);
  pair_scatter_kernel<<<(BB * PP + 255) / 256, 256, 0, stream>>>(
      p_lm, p_idx, pvmask, lnp_w, lnp_b, pb_w, bias);

  cond_kernel<<<dim3(MTOT / 16, 2), 128, 0, stream>>>(h_cond, tok, cond_w, cond_b,
                                                      t_emb, cond);
  ln_dual_kernel<<<MTOT / 8, 256, 0, stream>>>(cond, a1_snw, a2_snw, sn1, sn2);
  gates_kernel<<<MTOT / 16, 256, 0, stream>>>(cond, ag_w, ag_b, tg_w, tg_b, gA, gT);
  ln_dual_kernel<<<MTOT / 8, 256, 0, stream>>>(q, nullptr, nullptr, an, nullptr);
  adaln_kernel<<<MTOT / 16, 256, 0, stream>>>(sn1, an, a1_scw, a1_scb, a1_biw, qn);
  qkvg_kernel<<<MTOT / 16, 256, 0, stream>>>(qn, wq_w, wq_b, wk_w, wv_w, wg_w,
                                             Qb, Kb, Vb, Gb);
  attn_kernel<<<dim3(NAT / 64, BB), 256, 0, stream>>>(Qb, Kb, Vb, Gb, bias, pmask, ao);
  wo_kernel<<<MTOT / 16, 256, 0, stream>>>(ao, wo_w, q, gA, q2);
  ln_dual_kernel<<<MTOT / 8, 256, 0, stream>>>(q2, nullptr, nullptr, an2, nullptr);
  adaln_kernel<<<MTOT / 16, 256, 0, stream>>>(sn2, an2, a2_scw, a2_scb, a2_biw, qn2);
  swiglu1_kernel<<<dim3(MTOT / 16, 4), 256, 0, stream>>>(qn2, sw1, sw3, hid);
  final_kernel<<<MTOT / 16, 256, 0, stream>>>(hid, sw2, q2, gT, pmask, out);
}
